// GCNN_S2S_65695819759930
// MI455X (gfx1250) — matrix operand order repaired and checked by reading, not compiled, not run on hardware
//
#include <hip/hip_runtime.h>

typedef __attribute__((ext_vector_type(16))) __bf16 v16bf;
typedef __attribute__((ext_vector_type(8)))  __bf16 v8bf;
typedef __attribute__((ext_vector_type(8)))  float  v8f;
typedef unsigned int u32x4 __attribute__((ext_vector_type(4)));
typedef int          i32x4 __attribute__((ext_vector_type(4)));
typedef int          i32x8 __attribute__((ext_vector_type(8)));

#define NN   10000
#define EE   25000
#define BB   128
#define NF   32
#define EF   8
#define DD   64
#define STEPS 3
#define S2S   3

#if defined(__HIP_DEVICE_COMPILE__) && __has_builtin(__builtin_amdgcn_tensor_load_to_lds)
#define USE_TDM 1
#else
#define USE_TDM 0
#endif

static __device__ __forceinline__ float sigf(float x) { return 1.0f / (1.0f + __expf(-x)); }
static __device__ __forceinline__ int   f2ord(float f) { int i = __float_as_int(f); return i >= 0 ? i : (i ^ 0x7fffffff); }
static __device__ __forceinline__ float ord2f(int i)  { return __int_as_float(i >= 0 ? i : (i ^ 0x7fffffff)); }

// ---------------- generic helpers ----------------

__global__ void k_zero(float* __restrict__ p, int n) {
    int i = blockIdx.x * blockDim.x + threadIdx.x;
    if (i < n) p[i] = 0.0f;
}

__global__ void k_cvt_bf16(const float* __restrict__ s, __bf16* __restrict__ d, int n) {
    int i = blockIdx.x * blockDim.x + threadIdx.x;
    if (i < n) d[i] = (__bf16)s[i];
}

// ---------------- pack B[K,N] (f32 row-major) into wave32 WMMA B-fragment layout ----------------
// Bpack[(((nt*ktiles)+ks)*32 + lane)*16 + i] = (bf16)B[(ks*32 + kb(lane) + i + (i>=8 ? 8 : 0))*N + nt*16 + (lane&15)], kb = 8*(lane/16)
// so each lane's 16 B-fragment elements for (nt, ks) are 32 contiguous bytes.
__global__ void k_packB(const float* __restrict__ Bsrc, __bf16* __restrict__ Bpack, int K, int N)
{
    int t = blockIdx.x * blockDim.x + threadIdx.x;
    int ktiles = K >> 5;
    int total = (N >> 4) * ktiles * 32;
    if (t >= total) return;
    int lane = t & 31;
    int ksnt = t >> 5;
    int ks = ksnt % ktiles;
    int nt = ksnt / ktiles;
    int kb = (lane < 16) ? 0 : 8;
    int col = nt * 16 + (lane & 15);
    __bf16* dp = Bpack + (size_t)t * 16;
#pragma unroll
    for (int i = 0; i < 16; i++)
        dp[i] = (__bf16)Bsrc[(size_t)(ks * 32 + kb + i + (i >= 8 ? 8 : 0)) * N + col];
}

// ---------------- generic WMMA GEMM: C[M,N] = A[M,K](bf16) * Bpack + bias ----------------
// one wave per 16x16 tile; K multiple of 32, N multiple of 16; B pre-packed by k_packB.
__global__ void k_gemm_wmma(const __bf16* __restrict__ A, const __bf16* __restrict__ Bpack,
                            const float* __restrict__ bias, float* __restrict__ C,
                            __bf16* __restrict__ Cbf, int M, int N, int K, int relu)
{
    int lane = threadIdx.x & 31;
    int tn = blockIdx.x * 16;
    int tm = blockIdx.y * 16;
    int r  = lane & 15;
    bool hi = lane >= 16;
    int ktiles = K >> 5;

    int arow = tm + r; if (arow >= M) arow = M - 1;
    int ka = hi ? 8 : 0;
    const v16bf* Bv = (const v16bf*)Bpack + ((size_t)(tn >> 4) * ktiles) * 32 + lane;

    v8f acc = {0.f,0.f,0.f,0.f,0.f,0.f,0.f,0.f};
    for (int k0 = 0; k0 < K; k0 += 32) {
        v16bf a;
        const __bf16* Ap = A + (size_t)arow * K + k0;
#pragma unroll
        for (int i = 0; i < 8; i++) a[i]     = Ap[ka + i];
#pragma unroll
        for (int i = 0; i < 8; i++) a[8 + i] = Ap[16 + ka + i];
        v16bf b = Bv[(size_t)(k0 >> 5) * 32];
        acc = __builtin_amdgcn_wmma_f32_16x16x32_bf16(false, a, false, b, (short)0, acc, false, false);
    }
    int col = tn + (lane & 15);
    int rb  = hi ? 8 : 0;
    float bv = bias ? bias[col] : 0.0f;
#pragma unroll
    for (int v = 0; v < 8; v++) {
        int row = tm + rb + v;
        if (row < M) {
            float val = acc[v] + bv;
            if (relu) val = fmaxf(val, 0.0f);
            C[(size_t)row * N + col] = val;
            if (Cbf) Cbf[(size_t)row * N + col] = (__bf16)val;
        }
    }
}

// ---------------- edge MLP layer 1: h1 = relu(edge_attr @ We1 + be1) -> bf16 ----------------
__global__ void k_edge_l1(const float* __restrict__ ea, const float* __restrict__ We1,
                          const float* __restrict__ be1, __bf16* __restrict__ h1)
{
    int idx = blockIdx.x * blockDim.x + threadIdx.x;
    if (idx >= EE * 128) return;
    int e = idx >> 7, k = idx & 127;
    float s = be1[k];
#pragma unroll
    for (int j = 0; j < EF; j++) s += ea[e * EF + j] * We1[j * 128 + k];
    h1[idx] = (__bf16)fmaxf(s, 0.0f);
}

__global__ void k_deg(const int* __restrict__ dst, float* __restrict__ deg)
{
    int e = blockIdx.x * blockDim.x + threadIdx.x;
    if (e < EE) atomicAdd(&deg[dst[e]], 1.0f);
}

// ---------------- fused edge message kernel ----------------
// Tile of 16 edges: TDM stages the 16x128 h1 tile into LDS, then recompute
// ew[e,d,f] = (h1[e,:] @ We2[:, d*64+f]) + be2 chunk-by-chunk via WMMA and contract
// with out[src[e], d] on the fly; accumulate msg in LDS; scatter to agg[dst].
__global__ void k_edge_msg(const __bf16* __restrict__ h1, const __bf16* __restrict__ W2p,
                           const float* __restrict__ be2, const float* __restrict__ outf,
                           const int* __restrict__ src, const int* __restrict__ dst,
                           float* __restrict__ agg)
{
    __shared__ __align__(16) __bf16 h1s[16][128];   // 4KB staged A tile
    __shared__ float osrc[16][64];
    __shared__ float msg[16][64];

    int e0   = blockIdx.x * 16;
    int tid  = threadIdx.x;       // 0..127 (4 waves)
    int wave = tid >> 5;
    int lane = tid & 31;

    for (int i = tid; i < 16 * 64; i += 128) {
        int row = i >> 6, f = i & 63;
        int e = e0 + row; int ec = (e < EE) ? e : 0;
        osrc[row][f] = outf[(size_t)src[ec] * 64 + f];
        msg[row][f]  = 0.0f;
    }

#if USE_TDM
    if (wave == 0) {
        // D# group0: count=1 | lds_addr | global_addr | type=2
        unsigned lds_base = (unsigned)(uintptr_t)(&h1s[0][0]);
        unsigned long long ga = (unsigned long long)(uintptr_t)(h1 + (size_t)e0 * 128);
        u32x4 g0;
        g0[0] = 1u;
        g0[1] = lds_base;
        g0[2] = (unsigned)ga;
        g0[3] = ((unsigned)(ga >> 32) & 0x01FFFFFFu) | 0x80000000u;
        // D# group1: 1-D copy of 512 x 8-byte elements (16 rows x 128 bf16)
        i32x8 g1;
        g1[0] = (3 << 16);      // data_size = 8B
        g1[1] = (512 << 16);    // tensor_dim0[15:0] = 512 (bits 79:48)
        g1[2] = (1 << 16);      // tensor_dim0 hi = 0, tensor_dim1 = 1
        g1[3] = (512 << 16);    // tile_dim0 = 512 (bits 127:112)
        g1[4] = 1;              // tile_dim1 = 1, tile_dim2 = 0
        g1[5] = 512;            // tensor_dim0_stride = 512
        g1[6] = 0;
        g1[7] = 0;
        i32x4 z4 = {0, 0, 0, 0};
#if __clang_major__ >= 23
        i32x8 z8 = {0, 0, 0, 0, 0, 0, 0, 0};
        __builtin_amdgcn_tensor_load_to_lds(g0, g1, z4, z4, z8, 0);
#else
        __builtin_amdgcn_tensor_load_to_lds(g0, g1, z4, z4, 0);
#endif
#if __has_builtin(__builtin_amdgcn_s_wait_tensorcnt)
        __builtin_amdgcn_s_wait_tensorcnt(0);
#else
        asm volatile("s_wait_tensorcnt 0x0" ::: "memory");
#endif
    }
#else
    {   // fallback staging: 128 threads x 32B
        int row = tid >> 3, seg = tid & 7;
        int e = e0 + row; if (e >= EE) e = EE - 1;
        const __bf16* sp = h1 + (size_t)e * 128 + seg * 16;
        __bf16* dp = &h1s[row][seg * 16];
#pragma unroll
        for (int i = 0; i < 16; i++) dp[i] = sp[i];
    }
#endif
    __syncthreads();

    int r  = lane & 15;
    bool hi = lane >= 16;
    int ka = hi ? 8 : 0;
    int rb = hi ? 8 : 0;
    int bc = lane & 15;

    // A fragments for all 4 k-steps from LDS (aligned 16B vector loads)
    v16bf afrag[4];
#pragma unroll
    for (int ks = 0; ks < 4; ks++) {
        v8bf lo = *(const v8bf*)(&h1s[r][ks * 32 + ka]);
        v8bf hi8 = *(const v8bf*)(&h1s[r][ks * 32 + 16 + ka]);
#pragma unroll
        for (int i = 0; i < 8; i++) { afrag[ks][i] = lo[i]; afrag[ks][8 + i] = hi8[i]; }
    }

    float macc[4][8];
#pragma unroll
    for (int s = 0; s < 4; s++)
#pragma unroll
        for (int v = 0; v < 8; v++) macc[s][v] = 0.0f;

    const v16bf* W2v = (const v16bf*)W2p;
    for (int d = wave; d < 64; d += 4) {
        if (d + 4 < 64)  // pull next chunk of packed We2 toward the caches
            __builtin_prefetch(&W2v[(size_t)((d + 4) * 4) * 4 * 32 + lane], 0, 1);
#pragma unroll
        for (int s = 0; s < 4; s++) {
            int nt = d * 4 + s;
            v8f acc = {0.f,0.f,0.f,0.f,0.f,0.f,0.f,0.f};
#pragma unroll
            for (int ks = 0; ks < 4; ks++) {
                v16bf b = W2v[((size_t)nt * 4 + ks) * 32 + lane];
                acc = __builtin_amdgcn_wmma_f32_16x16x32_bf16(false, afrag[ks], false, b,
                                                              (short)0, acc, false, false);
            }
            float bias = be2[d * 64 + s * 16 + bc];
#pragma unroll
            for (int v = 0; v < 8; v++)
                macc[s][v] += (acc[v] + bias) * osrc[rb + v][d];
        }
    }
    // reduce partial messages across the 4 waves via LDS float atomics
#pragma unroll
    for (int s = 0; s < 4; s++)
#pragma unroll
        for (int v = 0; v < 8; v++)
            atomicAdd(&msg[rb + v][s * 16 + bc], macc[s][v]);
    __syncthreads();

    for (int i = tid; i < 16 * 64; i += 128) {
        int row = i >> 6, f = i & 63;
        int e = e0 + row;
        if (e < EE) atomicAdd(&agg[(size_t)dst[e] * 64 + f], msg[row][f]);
    }
}

// ---------------- m = relu(agg/deg + out@root + conv_b) -> bf16 ----------------
__global__ void k_mker(const float* __restrict__ agg, const float* __restrict__ deg,
                       const float* __restrict__ orb, const float* __restrict__ conv_b,
                       __bf16* __restrict__ mbf)
{
    int i = blockIdx.x * blockDim.x + threadIdx.x;
    if (i >= NN * 64) return;
    int n = i >> 6, f = i & 63;
    float v = fmaxf(agg[i] / fmaxf(deg[n], 1.0f) + orb[i] + conv_b[f], 0.0f);
    mbf[i] = (__bf16)v;
}

// ---------------- GRU cell elementwise ----------------
__global__ void k_gru(const float* __restrict__ gi, const float* __restrict__ gh,
                      float* __restrict__ outf, __bf16* __restrict__ outbf)
{
    int i = blockIdx.x * blockDim.x + threadIdx.x;
    if (i >= NN * 64) return;
    int n = i >> 6, f = i & 63;
    const float* gin = gi + (size_t)n * 192;
    const float* ghn = gh + (size_t)n * 192;
    float rg = sigf(gin[f]       + ghn[f]);
    float zg = sigf(gin[64 + f]  + ghn[64 + f]);
    float ng = tanhf(gin[128 + f] + rg * ghn[128 + f]);
    float hp = outf[i];
    float hn = (1.0f - zg) * ng + zg * hp;
    outf[i]  = hn;
    outbf[i] = (__bf16)hn;
}

// ---------------- Set2Set ----------------
__global__ void k_lstm(const float* __restrict__ q_star, float* __restrict__ hl,
                       float* __restrict__ cl, const float* __restrict__ Wli,
                       const float* __restrict__ bli, const float* __restrict__ Wlh,
                       const float* __restrict__ blh)
{
    __shared__ float g[256];
    __shared__ float qs[128];
    __shared__ float hs[64];
    int b = blockIdx.x, t = threadIdx.x;
    if (t < 128) qs[t] = q_star[b * 128 + t];
    if (t < 64)  hs[t] = hl[b * 64 + t];
    __syncthreads();
    float s = bli[t] + blh[t];
    for (int k = 0; k < 128; k++) s += qs[k] * Wli[k * 256 + t];
    for (int k = 0; k < 64;  k++) s += hs[k] * Wlh[k * 256 + t];
    g[t] = s;
    __syncthreads();
    if (t < 64) {
        float ig = sigf(g[t]);
        float fg = sigf(g[64 + t]);
        float gg = tanhf(g[128 + t]);
        float og = sigf(g[192 + t]);
        float c  = fg * cl[b * 64 + t] + ig * gg;
        cl[b * 64 + t] = c;
        hl[b * 64 + t] = og * tanhf(c);
    }
}

__global__ void k_s2s_init(float* __restrict__ r_read, float* __restrict__ asum,
                           int* __restrict__ emax_i)
{
    int t = blockIdx.x * blockDim.x + threadIdx.x;
    if (t < BB * 64) r_read[t] = 0.0f;
    if (t < BB) { asum[t] = 0.0f; emax_i[t] = (int)0x80000000; }
}

__global__ void k_elogit(const float* __restrict__ outf, const float* __restrict__ hl,
                         const int* __restrict__ batch, float* __restrict__ elog,
                         int* __restrict__ emax_i)
{
    int n = blockIdx.x * blockDim.x + threadIdx.x;
    if (n >= NN) return;
    int b = batch[n];
    const float* o = outf + (size_t)n * 64;
    const float* q = hl + (size_t)b * 64;
    float s = 0.0f;
#pragma unroll
    for (int f = 0; f < 64; f++) s += o[f] * q[f];
    elog[n] = s;
    atomicMax(&emax_i[b], f2ord(s));
}

__global__ void k_expsum(const float* __restrict__ elog, const int* __restrict__ batch,
                         const int* __restrict__ emax_i, float* __restrict__ aatt,
                         float* __restrict__ asum)
{
    int n = blockIdx.x * blockDim.x + threadIdx.x;
    if (n >= NN) return;
    int b = batch[n];
    float av = __expf(elog[n] - ord2f(emax_i[b]));
    aatt[n] = av;
    atomicAdd(&asum[b], av);
}

__global__ void k_wread(const float* __restrict__ aatt, const float* __restrict__ asum,
                        const int* __restrict__ batch, const float* __restrict__ outf,
                        float* __restrict__ r_read)
{
    int i = blockIdx.x * blockDim.x + threadIdx.x;
    if (i >= NN * 64) return;
    int n = i >> 6, f = i & 63;
    int b = batch[n];
    float w = aatt[n] / fmaxf(asum[b], 1e-12f);
    atomicAdd(&r_read[b * 64 + f], w * outf[i]);
}

__global__ void k_qstar(const float* __restrict__ hl, const float* __restrict__ r_read,
                        float* __restrict__ q_star)
{
    int t = blockIdx.x * blockDim.x + threadIdx.x;
    if (t >= BB * 128) return;
    int b = t >> 7, j = t & 127;
    q_star[t] = (j < 64) ? hl[b * 64 + j] : r_read[b * 64 + (j - 64)];
}

// ---------------- head ----------------
__global__ void k_head1(const float* __restrict__ q_star, const float* __restrict__ Wr1,
                        const float* __restrict__ br1, float* __restrict__ y)
{
    __shared__ float qs[128];
    int b = blockIdx.x, f = threadIdx.x;
    qs[f] = q_star[b * 128 + f];
    qs[64 + f] = q_star[b * 128 + 64 + f];
    __syncthreads();
    float s = br1[f];
    for (int k = 0; k < 128; k++) s += qs[k] * Wr1[k * 64 + f];
    y[b * 64 + f] = s;
}

__global__ void k_head2(const float* __restrict__ y, float* __restrict__ mud,
                        float* __restrict__ rstd)
{
    int f = threadIdx.x;
    float mu = 0.0f;
    for (int b = 0; b < BB; b++) mu += y[b * 64 + f];
    mu /= (float)BB;
    float var = 0.0f;
    for (int b = 0; b < BB; b++) { float d = y[b * 64 + f] - mu; var += d * d; }
    var /= (float)BB;
    mud[f] = mu;
    rstd[f] = rsqrtf(var + 1e-5f);
}

__global__ void k_head3(const float* __restrict__ y, const float* __restrict__ mud,
                        const float* __restrict__ rstd, const float* __restrict__ gamma,
                        const float* __restrict__ beta, const float* __restrict__ Wr2,
                        const float* __restrict__ br2, float* __restrict__ outp)
{
    __shared__ float red[64];
    int b = blockIdx.x, f = threadIdx.x;
    float v = (y[b * 64 + f] - mud[f]) * rstd[f] * gamma[f] + beta[f];
    v = fmaxf(v, 0.0f);
    red[f] = v * Wr2[f];
    __syncthreads();
    for (int s = 32; s > 0; s >>= 1) {
        if (f < s) red[f] += red[f + s];
        __syncthreads();
    }
    if (f == 0) outp[b] = red[0] + br2[0];
}

// ---------------- launch ----------------
extern "C" void kernel_launch(void* const* d_in, const int* in_sizes, int n_in,
                              void* d_out, int out_size, void* d_ws, size_t ws_size,
                              hipStream_t stream)
{
    (void)in_sizes; (void)n_in; (void)out_size; (void)ws_size;
    const float* x      = (const float*)d_in[0];
    const int*   eidx   = (const int*)d_in[1];
    const float* eattr  = (const float*)d_in[2];
    const int*   batch  = (const int*)d_in[3];
    const float* W0     = (const float*)d_in[4];
    const float* b0     = (const float*)d_in[5];
    const float* We1    = (const float*)d_in[6];
    const float* be1    = (const float*)d_in[7];
    const float* We2    = (const float*)d_in[8];
    const float* be2    = (const float*)d_in[9];
    const float* root   = (const float*)d_in[10];
    const float* conv_b = (const float*)d_in[11];
    const float* Wi     = (const float*)d_in[12];
    const float* bi     = (const float*)d_in[13];
    const float* Wh     = (const float*)d_in[14];
    const float* bh     = (const float*)d_in[15];
    const float* Wli    = (const float*)d_in[16];
    const float* bli    = (const float*)d_in[17];
    const float* Wlh    = (const float*)d_in[18];
    const float* blh    = (const float*)d_in[19];
    const float* Wr1    = (const float*)d_in[20];
    const float* br1    = (const float*)d_in[21];
    const float* gamma  = (const float*)d_in[22];
    const float* beta   = (const float*)d_in[23];
    const float* Wr2    = (const float*)d_in[24];
    const float* br2    = (const float*)d_in[25];
    const int* src = eidx;
    const int* dst = eidx + EE;
    float* outp = (float*)d_out;

    char* w = (char*)d_ws;
    size_t off = 0;
    auto alloc = [&](size_t bytes) -> void* {
        void* p = w + off;
        off = (off + bytes + 255) & ~(size_t)255;
        return p;
    };
    float*  outf   = (float*)alloc((size_t)NN * 64 * 4);
    __bf16* outbf  = (__bf16*)alloc((size_t)NN * 64 * 2);
    __bf16* mbf    = (__bf16*)alloc((size_t)NN * 64 * 2);
    float*  agg    = (float*)alloc((size_t)NN * 64 * 4);
    float*  orb    = (float*)alloc((size_t)NN * 64 * 4);
    float*  gi     = (float*)alloc((size_t)NN * 192 * 4);
    float*  gh     = (float*)alloc((size_t)NN * 192 * 4);
    __bf16* h1bf   = (__bf16*)alloc((size_t)EE * 128 * 2);
    float*  deg    = (float*)alloc((size_t)NN * 4);
    __bf16* xbf    = (__bf16*)alloc((size_t)NN * NF * 2);
    __bf16* W0p    = (__bf16*)alloc(NF * 64 * 2);        // packed fragments (same elem count)
    __bf16* We2p   = (__bf16*)alloc(128 * 4096 * 2);
    __bf16* rootp  = (__bf16*)alloc(64 * 64 * 2);
    __bf16* Wip    = (__bf16*)alloc(64 * 192 * 2);
    __bf16* Whp    = (__bf16*)alloc(64 * 192 * 2);
    float*  elog   = (float*)alloc((size_t)NN * 4);
    float*  aatt   = (float*)alloc((size_t)NN * 4);
    int*    emax_i = (int*)alloc(BB * 4);
    float*  asum   = (float*)alloc(BB * 4);
    float*  r_read = (float*)alloc(BB * 64 * 4);
    float*  q_star = (float*)alloc(BB * 128 * 4);
    float*  hl     = (float*)alloc(BB * 64 * 4);
    float*  cl     = (float*)alloc(BB * 64 * 4);
    float*  y      = (float*)alloc(BB * 64 * 4);
    float*  mud    = (float*)alloc(64 * 4);
    float*  rstd   = (float*)alloc(64 * 4);

    auto cdiv = [](int a, int b) { return (a + b - 1) / b; };

    // input conversion + weight packing (fuses f32->bf16 with fragment layout)
    k_cvt_bf16<<<cdiv(NN * NF, 256), 256, 0, stream>>>(x, xbf, NN * NF);
    k_packB<<<cdiv((64 / 16) * (32 / 32) * 32, 128), 128, 0, stream>>>(W0, W0p, 32, 64);
    k_packB<<<cdiv((4096 / 16) * (128 / 32) * 32, 256), 256, 0, stream>>>(We2, We2p, 128, 4096);
    k_packB<<<cdiv((64 / 16) * (64 / 32) * 32, 128), 128, 0, stream>>>(root, rootp, 64, 64);
    k_packB<<<cdiv((192 / 16) * (64 / 32) * 32, 128), 128, 0, stream>>>(Wi, Wip, 64, 192);
    k_packB<<<cdiv((192 / 16) * (64 / 32) * 32, 128), 128, 0, stream>>>(Wh, Whp, 64, 192);

    // dense0: out = relu(x@W0 + b0)
    {
        dim3 g(64 / 16, cdiv(NN, 16));
        k_gemm_wmma<<<g, 32, 0, stream>>>(xbf, W0p, b0, outf, outbf, NN, 64, NF, 1);
    }
    // edge MLP layer 1
    k_edge_l1<<<cdiv(EE * 128, 256), 256, 0, stream>>>(eattr, We1, be1, h1bf);
    // degrees
    k_zero<<<cdiv(NN, 256), 256, 0, stream>>>(deg, NN);
    k_deg<<<cdiv(EE, 256), 256, 0, stream>>>(dst, deg);

    for (int step = 0; step < STEPS; step++) {
        k_zero<<<cdiv(NN * 64, 256), 256, 0, stream>>>(agg, NN * 64);
        k_edge_msg<<<cdiv(EE, 16), 128, 0, stream>>>(h1bf, We2p, be2, outf, src, dst, agg);
        {
            dim3 g(64 / 16, cdiv(NN, 16));
            k_gemm_wmma<<<g, 32, 0, stream>>>(outbf, rootp, nullptr, orb, nullptr, NN, 64, 64, 0);
        }
        k_mker<<<cdiv(NN * 64, 256), 256, 0, stream>>>(agg, deg, orb, conv_b, mbf);
        {
            dim3 g(192 / 16, cdiv(NN, 16));
            k_gemm_wmma<<<g, 32, 0, stream>>>(mbf, Wip, bi, gi, nullptr, NN, 192, 64, 0);
            k_gemm_wmma<<<g, 32, 0, stream>>>(outbf, Whp, bh, gh, nullptr, NN, 192, 64, 0);
        }
        k_gru<<<cdiv(NN * 64, 256), 256, 0, stream>>>(gi, gh, outf, outbf);
    }

    // Set2Set
    k_zero<<<cdiv(BB * 128, 256), 256, 0, stream>>>(q_star, BB * 128);
    k_zero<<<cdiv(BB * 64, 256), 256, 0, stream>>>(hl, BB * 64);
    k_zero<<<cdiv(BB * 64, 256), 256, 0, stream>>>(cl, BB * 64);
    for (int t = 0; t < S2S; t++) {
        k_lstm<<<BB, 256, 0, stream>>>(q_star, hl, cl, Wli, bli, Wlh, blh);
        k_s2s_init<<<cdiv(BB * 64, 256), 256, 0, stream>>>(r_read, asum, emax_i);
        k_elogit<<<cdiv(NN, 256), 256, 0, stream>>>(outf, hl, batch, elog, emax_i);
        k_expsum<<<cdiv(NN, 256), 256, 0, stream>>>(elog, batch, emax_i, aatt, asum);
        k_wread<<<cdiv(NN * 64, 256), 256, 0, stream>>>(aatt, asum, batch, outf, r_read);
        k_qstar<<<cdiv(BB * 128, 256), 256, 0, stream>>>(hl, r_read, q_star);
    }

    // regression head
    k_head1<<<BB, 64, 0, stream>>>(q_star, Wr1, br1, y);
    k_head2<<<1, 64, 0, stream>>>(y, mud, rstd);
    k_head3<<<BB, 64, 0, stream>>>(y, mud, rstd, gamma, beta, Wr2, br2, outp);
}
